// FixedPointSolver_72885595013516
// MI455X (gfx1250) — matrix operand order repaired and checked by reading, not compiled, not run on hardware
//
#include <hip/hip_runtime.h>
#include <hip/hip_bf16.h>

// DEQ fixed-point solver (Anderson acceleration, m=5, beta=1, 20 iters) +
// Hutchinson jac_reg, for B=d=2048, on gfx1250 via V_WMMA_F32_16X16X32_F16
// with double-buffered async global->LDS staging (ASYNCcnt path) when available.

#define DDIM 2048
#define NTOT (2048 * 2048)
#define GRAM_BLOCKS 256

typedef __attribute__((ext_vector_type(16))) _Float16 v16h;
typedef __attribute__((ext_vector_type(8)))  _Float16 v8h;
typedef __attribute__((ext_vector_type(8)))  float    v8f;
typedef __attribute__((ext_vector_type(4)))  int      v4i;

union Frag { v16h v; v8h h[2]; };

#if defined(__has_builtin)
#if __has_builtin(__builtin_amdgcn_global_load_async_to_lds_b128)
#define HAVE_ASYNC_LDS 1
#else
#define HAVE_ASYNC_LDS 0
#endif
#else
#define HAVE_ASYNC_LDS 0
#endif

#if HAVE_ASYNC_LDS
typedef __attribute__((address_space(1))) v4i gv4i_t;
typedef __attribute__((address_space(3))) v4i lv4i_t;
__device__ __forceinline__ void async_cp16(const _Float16* g, _Float16* l) {
  __builtin_amdgcn_global_load_async_to_lds_b128((gv4i_t*)g, (lv4i_t*)l, 0, 0);
}
template <int N> __device__ __forceinline__ void wait_async() {
#if __has_builtin(__builtin_amdgcn_s_wait_asynccnt)
  __builtin_amdgcn_s_wait_asynccnt(N);
#else
  asm volatile("s_wait_asynccnt %0" ::"i"(N));
#endif
}
#endif

__device__ __forceinline__ float fast_tanh(float x) {
  float a = __builtin_fminf(__builtin_fabsf(x), 15.0f);
  float e = __expf(-2.0f * a);
  float t = __fdividef(1.0f - e, 1.0f + e);
  return __builtin_copysignf(t, x);
}

// ---------------------------------------------------------------------------
// One-time: Wt[n][k] = (f16) W[k][n]   (tiled transpose through LDS)
// grid (64,64), block (32,8)
// ---------------------------------------------------------------------------
__global__ void __launch_bounds__(256) wtrans_kernel(const float* __restrict__ W,
                                                     _Float16* __restrict__ Wt) {
  __shared__ _Float16 tile[32][33];
  const int bk = blockIdx.x * 32;   // K dim of W (rows)
  const int bn = blockIdx.y * 32;   // N dim of W (cols)
  const int tx = threadIdx.x, ty = threadIdx.y;
  #pragma unroll
  for (int r = 0; r < 32; r += 8)
    tile[ty + r][tx] = (_Float16)W[(size_t)(bk + ty + r) * DDIM + (bn + tx)];
  __syncthreads();
  #pragma unroll
  for (int r = 0; r < 32; r += 8)
    Wt[(size_t)(bn + ty + r) * DDIM + (bk + tx)] = tile[tx][ty + r];
}

// ---------------------------------------------------------------------------
// Init: F0 = tanh(u); X0 = 0; X1 = F0; Z16 = f16(F0)
// ---------------------------------------------------------------------------
__global__ void __launch_bounds__(256) init_kernel(const float* __restrict__ u,
                                                   float* __restrict__ F0,
                                                   float* __restrict__ X0,
                                                   float* __restrict__ X1,
                                                   _Float16* __restrict__ Z16) {
  int i = blockIdx.x * 256 + threadIdx.x;
  float t = fast_tanh(u[i]);
  F0[i] = t; X0[i] = 0.0f; X1[i] = t; Z16[i] = (_Float16)t;
}

// ---------------------------------------------------------------------------
// Core GEMM: Y = tanh(A16 (MxK,f16) @ W + bias) using Wt (NxK) as B source.
// Block tile 128x128, 8 waves (2x4), wave tile 64x32, K step 32,
// double-buffered LDS with async global->LDS staging.
// grid (16,16), block 256.
// ---------------------------------------------------------------------------
__global__ void __launch_bounds__(256) gemm_tanh_kernel(
    const _Float16* __restrict__ A16,   // (D,D) row-major activations
    const _Float16* __restrict__ Bt16,  // (D,D) row-major = W^T (n-major)
    const float*    __restrict__ bias,  // u
    float*          __restrict__ Yf,
    _Float16*       __restrict__ Y16) {
  __shared__ _Float16 As[2][128 * 40];
  __shared__ _Float16 Bs[2][128 * 40];
  const int tid  = threadIdx.x;
  const int lane = tid & 31, wid = tid >> 5;
  const int lo = lane & 15, hi = lane >> 4;
  const int bm0 = blockIdx.y * 128;
  const int bn0 = blockIdx.x * 128;
  const int wm0 = (wid >> 2) * 64;
  const int wn0 = (wid & 3) * 32;

  // staging: thread owns v8h chunks at rows r0 and r0+64, fixed col group c8
  const int r0 = tid >> 2, c8 = (tid & 3) << 3;
  const _Float16* gA = A16  + (size_t)(bm0 + r0) * DDIM + c8;
  const _Float16* gB = Bt16 + (size_t)(bn0 + r0) * DDIM + c8;
  const int lofs = r0 * 40 + c8;

  v8f acc[4][2] = {};

  auto stage = [&](int buf, int kb) {
#if HAVE_ASYNC_LDS
    async_cp16(gA + kb,                        &As[buf][lofs]);
    async_cp16(gA + kb + (size_t)64 * DDIM,    &As[buf][lofs + 64 * 40]);
    async_cp16(gB + kb,                        &Bs[buf][lofs]);
    async_cp16(gB + kb + (size_t)64 * DDIM,    &Bs[buf][lofs + 64 * 40]);
#else
    *(v8h*)&As[buf][lofs]           = *(const v8h*)(gA + kb);
    *(v8h*)&As[buf][lofs + 64 * 40] = *(const v8h*)(gA + kb + (size_t)64 * DDIM);
    *(v8h*)&Bs[buf][lofs]           = *(const v8h*)(gB + kb);
    *(v8h*)&Bs[buf][lofs + 64 * 40] = *(const v8h*)(gB + kb + (size_t)64 * DDIM);
#endif
  };

  stage(0, 0);
  for (int kb = 0; kb < DDIM; kb += 32) {
    const int cur = (kb >> 5) & 1;
    if (kb + 32 < DDIM) {
      stage(cur ^ 1, kb + 32);   // prefetch next tile into other buffer
#if HAVE_ASYNC_LDS
      wait_async<4>();           // current tile landed; next batch in flight
#endif
    } else {
#if HAVE_ASYNC_LDS
      wait_async<0>();
#endif
    }
    __syncthreads();

    const _Float16* Ab = As[cur];
    const _Float16* Bb = Bs[cur];
    Frag af[4], bf[2];
    #pragma unroll
    for (int mi = 0; mi < 4; ++mi) {
      const _Float16* ap = Ab + (wm0 + mi * 16 + lo) * 40 + hi * 8;
      af[mi].h[0] = *(const v8h*)ap;          // K = c0..c0+7
      af[mi].h[1] = *(const v8h*)(ap + 16);   // K = 16+c0..16+c0+7
    }
    #pragma unroll
    for (int ni = 0; ni < 2; ++ni) {
      const _Float16* bp = Bb + (wn0 + ni * 16 + lo) * 40 + hi * 8;
      bf[ni].h[0] = *(const v8h*)bp;          // K = hi*8 .. +7
      bf[ni].h[1] = *(const v8h*)(bp + 16);   // K = 16+hi*8 .. +7
    }
    #pragma unroll
    for (int mi = 0; mi < 4; ++mi)
      #pragma unroll
      for (int ni = 0; ni < 2; ++ni)
        acc[mi][ni] = __builtin_amdgcn_wmma_f32_16x16x32_f16(
            false, af[mi].v, false, bf[ni].v, (short)0, acc[mi][ni], false, false);
    __syncthreads();  // everyone done reading 'cur' before it is restaged
  }

  #pragma unroll
  for (int mi = 0; mi < 4; ++mi) {
    #pragma unroll
    for (int ni = 0; ni < 2; ++ni) {
      const int col  = bn0 + wn0 + ni * 16 + lo;
      const int row0 = bm0 + wm0 + mi * 16 + hi * 8;
      #pragma unroll
      for (int r = 0; r < 8; ++r) {
        size_t off = (size_t)(row0 + r) * DDIM + col;
        float t = fast_tanh(acc[mi][ni][r] + bias[off]);
        Yf[off]  = t;
        Y16[off] = (_Float16)t;
      }
    }
  }
}

// ---------------------------------------------------------------------------
// Gram: partial sums of G_i . G_j  (G_i = F_i - X_i), pairs i<=j, per block.
// ---------------------------------------------------------------------------
__global__ void __launch_bounds__(256) gram_kernel(const float* __restrict__ F,
                                                   const float* __restrict__ X,
                                                   float* __restrict__ part, int nk) {
  const int n = NTOT;
  float acc[15];
  #pragma unroll
  for (int p = 0; p < 15; ++p) acc[p] = 0.0f;
  for (int idx = blockIdx.x * 256 + threadIdx.x; idx < n; idx += GRAM_BLOCKS * 256) {
    float g[5];
    #pragma unroll
    for (int i = 0; i < 5; ++i)
      g[i] = (i < nk) ? (F[(size_t)i * n + idx] - X[(size_t)i * n + idx]) : 0.0f;
    int p = 0;
    #pragma unroll
    for (int i = 0; i < 5; ++i)
      #pragma unroll
      for (int j = i; j < 5; ++j) { acc[p] += g[i] * g[j]; ++p; }
  }
  __shared__ float red[256];
  const int tid = threadIdx.x;
  #pragma unroll
  for (int p = 0; p < 15; ++p) {
    red[tid] = acc[p]; __syncthreads();
    for (int s = 128; s > 0; s >>= 1) { if (tid < s) red[tid] += red[tid + s]; __syncthreads(); }
    if (tid == 0) part[blockIdx.x * 16 + p] = red[0];
    __syncthreads();
  }
}

// ---------------------------------------------------------------------------
// Solve (H + lam I) a = 1, normalize. Single block; pair reduce then 1 thread.
// ---------------------------------------------------------------------------
__global__ void __launch_bounds__(64) solve_kernel(const float* __restrict__ part,
                                                   float* __restrict__ alpha, int nk) {
  __shared__ float pr[15];
  const int tid = threadIdx.x;
  if (tid < 15) {
    float s = 0.0f;
    for (int b = 0; b < GRAM_BLOCKS; ++b) s += part[b * 16 + tid];
    pr[tid] = s;
  }
  __syncthreads();
  if (tid == 0) {
    float H[5][5], rhs[5], a[5];
    int p = 0;
    for (int i = 0; i < 5; ++i)
      for (int j = i; j < 5; ++j) { H[i][j] = pr[p]; H[j][i] = pr[p]; ++p; }
    for (int i = 0; i < nk; ++i) { H[i][i] += 1e-4f; rhs[i] = 1.0f; }
    for (int i = 0; i < nk; ++i) {             // Gaussian elimination (SPD)
      float inv = __fdividef(1.0f, H[i][i]);
      for (int j = i + 1; j < nk; ++j) {
        float f = H[j][i] * inv;
        for (int c = i; c < nk; ++c) H[j][c] -= f * H[i][c];
        rhs[j] -= f * rhs[i];
      }
    }
    for (int i = nk - 1; i >= 0; --i) {
      float s = rhs[i];
      for (int j = i + 1; j < nk; ++j) s -= H[i][j] * a[j];
      a[i] = __fdividef(s, H[i][i]);
    }
    float tot = 0.0f;
    for (int i = 0; i < nk; ++i) tot += a[i];
    float it = __fdividef(1.0f, tot);
    for (int i = 0; i < 5; ++i) alpha[i] = (i < nk) ? a[i] * it : 0.0f;
  }
}

// ---------------------------------------------------------------------------
// Mix: xk = sum_i alpha_i F_i  (beta = 1) -> X[slot] f32 and Z16 f16
// ---------------------------------------------------------------------------
__global__ void __launch_bounds__(256) mix_kernel(const float* __restrict__ F,
                                                  const float* __restrict__ alpha, int nk,
                                                  float* __restrict__ Xslot,
                                                  _Float16* __restrict__ Z16) {
  int idx = blockIdx.x * 256 + threadIdx.x;
  float s = 0.0f;
  #pragma unroll
  for (int i = 0; i < 5; ++i)
    if (i < nk) s += alpha[i] * F[(size_t)i * NTOT + idx];
  Xslot[idx] = s;
  Z16[idx] = (_Float16)s;
}

// ---------------------------------------------------------------------------
// eps ~ N(0,1) via hash + Box-Muller (deterministic), stored as f16.
// ---------------------------------------------------------------------------
__device__ __forceinline__ unsigned hash32(unsigned x) {
  x ^= x >> 16; x *= 0x7feb352du; x ^= x >> 15; x *= 0x846ca68bu; x ^= x >> 16;
  return x;
}
__global__ void __launch_bounds__(256) eps_kernel(_Float16* __restrict__ E) {
  int i = blockIdx.x * 256 + threadIdx.x;
  unsigned h1 = hash32(0x9e3779b9u ^ (unsigned)i);
  unsigned h2 = hash32(0x85ebca6bu + (unsigned)i);
  float u1 = ((float)(h1 >> 8) + 1.0f) * (1.0f / 16777217.0f);
  float u2 = (float)(h2 >> 8) * (1.0f / 16777216.0f);
  float r = __fsqrt_rn(-2.0f * __logf(u1));
  E[i] = (_Float16)(r * __cosf(6.2831853f * u2));
}

// ---------------------------------------------------------------------------
// Fused JVP: s = tanh(x*W+u), t = eps W, partial-sum of ((1-s^2) t)^2 per block
// Block tile 64x128, wave tile 32x32, shared B fragment for both GEMMs,
// double-buffered async staging.
// grid (16,32), block 256.
// ---------------------------------------------------------------------------
__global__ void __launch_bounds__(256) jvp_kernel(
    const _Float16* __restrict__ Xs16, const _Float16* __restrict__ E16,
    const _Float16* __restrict__ Bt16, const float* __restrict__ bias,
    float* __restrict__ part) {
  __shared__ _Float16 Axs[2][64 * 40];
  __shared__ _Float16 Aes[2][64 * 40];
  __shared__ _Float16 Bls[2][128 * 40];
  __shared__ float red[256];
  const int tid  = threadIdx.x;
  const int lane = tid & 31, wid = tid >> 5;
  const int lo = lane & 15, hi = lane >> 4;
  const int bm0 = blockIdx.y * 64;
  const int bn0 = blockIdx.x * 128;
  const int wm0 = (wid >> 2) * 32;
  const int wn0 = (wid & 3) * 32;

  const int r0 = tid >> 2, c8 = (tid & 3) << 3;   // r0 in 0..63
  const _Float16* gAx = Xs16 + (size_t)(bm0 + r0) * DDIM + c8;
  const _Float16* gAe = E16  + (size_t)(bm0 + r0) * DDIM + c8;
  const _Float16* gB  = Bt16 + (size_t)(bn0 + r0) * DDIM + c8;
  const int lofs = r0 * 40 + c8;

  v8f accS[2][2] = {}, accT[2][2] = {};

  auto stage = [&](int buf, int kb) {
#if HAVE_ASYNC_LDS
    async_cp16(gAx + kb,                     &Axs[buf][lofs]);
    async_cp16(gAe + kb,                     &Aes[buf][lofs]);
    async_cp16(gB  + kb,                     &Bls[buf][lofs]);
    async_cp16(gB  + kb + (size_t)64 * DDIM, &Bls[buf][lofs + 64 * 40]);
#else
    *(v8h*)&Axs[buf][lofs]           = *(const v8h*)(gAx + kb);
    *(v8h*)&Aes[buf][lofs]           = *(const v8h*)(gAe + kb);
    *(v8h*)&Bls[buf][lofs]           = *(const v8h*)(gB + kb);
    *(v8h*)&Bls[buf][lofs + 64 * 40] = *(const v8h*)(gB + kb + (size_t)64 * DDIM);
#endif
  };

  stage(0, 0);
  for (int kb = 0; kb < DDIM; kb += 32) {
    const int cur = (kb >> 5) & 1;
    if (kb + 32 < DDIM) {
      stage(cur ^ 1, kb + 32);
#if HAVE_ASYNC_LDS
      wait_async<4>();
#endif
    } else {
#if HAVE_ASYNC_LDS
      wait_async<0>();
#endif
    }
    __syncthreads();

    Frag ax[2], ae[2], bf[2];
    #pragma unroll
    for (int mi = 0; mi < 2; ++mi) {
      const _Float16* px = &Axs[cur][(wm0 + mi * 16 + lo) * 40 + hi * 8];
      const _Float16* pe = &Aes[cur][(wm0 + mi * 16 + lo) * 40 + hi * 8];
      ax[mi].h[0] = *(const v8h*)px;  ax[mi].h[1] = *(const v8h*)(px + 16);
      ae[mi].h[0] = *(const v8h*)pe;  ae[mi].h[1] = *(const v8h*)(pe + 16);
    }
    #pragma unroll
    for (int ni = 0; ni < 2; ++ni) {
      const _Float16* bp = &Bls[cur][(wn0 + ni * 16 + lo) * 40 + hi * 8];
      bf[ni].h[0] = *(const v8h*)bp;  bf[ni].h[1] = *(const v8h*)(bp + 16);
    }
    #pragma unroll
    for (int mi = 0; mi < 2; ++mi)
      #pragma unroll
      for (int ni = 0; ni < 2; ++ni) {
        accS[mi][ni] = __builtin_amdgcn_wmma_f32_16x16x32_f16(
            false, ax[mi].v, false, bf[ni].v, (short)0, accS[mi][ni], false, false);
        accT[mi][ni] = __builtin_amdgcn_wmma_f32_16x16x32_f16(
            false, ae[mi].v, false, bf[ni].v, (short)0, accT[mi][ni], false, false);
      }
    __syncthreads();
  }

  float lsum = 0.0f;
  #pragma unroll
  for (int mi = 0; mi < 2; ++mi)
    #pragma unroll
    for (int ni = 0; ni < 2; ++ni) {
      const int col  = bn0 + wn0 + ni * 16 + lo;
      const int row0 = bm0 + wm0 + mi * 16 + hi * 8;
      #pragma unroll
      for (int r = 0; r < 8; ++r) {
        size_t off = (size_t)(row0 + r) * DDIM + col;
        float s = fast_tanh(accS[mi][ni][r] + bias[off]);
        float g = (1.0f - s * s) * accT[mi][ni][r];
        lsum += g * g;
      }
    }
  red[tid] = lsum; __syncthreads();
  for (int s = 128; s > 0; s >>= 1) { if (tid < s) red[tid] += red[tid + s]; __syncthreads(); }
  if (tid == 0) part[blockIdx.y * gridDim.x + blockIdx.x] = red[0];
}

__global__ void __launch_bounds__(256) finish_kernel(const float* __restrict__ part,
                                                     float* __restrict__ out) {
  __shared__ float red[256];
  const int tid = threadIdx.x;
  red[tid] = part[tid] + part[tid + 256];
  __syncthreads();
  for (int s = 128; s > 0; s >>= 1) { if (tid < s) red[tid] += red[tid + s]; __syncthreads(); }
  if (tid == 0) out[0] = red[0] / (float)NTOT;
}

// ---------------------------------------------------------------------------
extern "C" void kernel_launch(void* const* d_in, const int* in_sizes, int n_in,
                              void* d_out, int out_size, void* d_ws, size_t ws_size,
                              hipStream_t stream) {
  const float* Wf = (const float*)d_in[1];
  const float* U  = (const float*)d_in[2];
  const size_t NT = (size_t)NTOT;

  char* w = (char*)d_ws;
  _Float16* Wt    = (_Float16*)w; w += NT * 2;      // W^T, f16
  _Float16* Z16   = (_Float16*)w; w += NT * 2;      // current GEMM input (f16)
  _Float16* Yt16  = (_Float16*)w; w += NT * 2;      // f16 GEMM output (x*16 at end)
  _Float16* E16   = (_Float16*)w; w += NT * 2;      // Hutchinson probe (f16)
  float* Xh       = (float*)w;    w += NT * 4 * 5;  // Anderson X history
  float* Fh       = (float*)w;    w += NT * 4 * 5;  // Anderson F history
  float* gpart    = (float*)w;    w += GRAM_BLOCKS * 16 * 4;
  float* alpha    = (float*)w;    w += 64;
  float* jpart    = (float*)w;    w += 512 * 4;

  float* out_x = (float*)d_out;
  float* out_j = out_x + NT;

  const dim3 gemm_grid(16, 16), blk(256);
  const int ew_blocks = NTOT / 256;

  // W -> W^T (f16), once
  wtrans_kernel<<<dim3(64, 64), dim3(32, 8), 0, stream>>>(Wf, Wt);

  // k=0,1: F0 = tanh(u); X0 = 0; X1 = F0; F1 = g(F0)
  init_kernel<<<ew_blocks, blk, 0, stream>>>(U, Fh, Xh, Xh + NT, Z16);
  gemm_tanh_kernel<<<gemm_grid, blk, 0, stream>>>(Z16, Wt, U, Fh + NT, Yt16);

  // k = 2..19: Anderson step + fixed-point eval
  for (int k = 2; k < 20; ++k) {
    const int nk = (k < 5) ? k : 5;
    const int slot = k % 5;
    gram_kernel<<<GRAM_BLOCKS, blk, 0, stream>>>(Fh, Xh, gpart, nk);
    solve_kernel<<<1, 64, 0, stream>>>(gpart, alpha, nk);
    mix_kernel<<<ew_blocks, blk, 0, stream>>>(Fh, alpha, nk, Xh + (size_t)slot * NT, Z16);
    float* yout = (k == 19) ? out_x : (Fh + (size_t)slot * NT);
    gemm_tanh_kernel<<<gemm_grid, blk, 0, stream>>>(Z16, Wt, U, yout, Yt16);
  }

  // jac_reg: eps, fused dual-GEMM JVP, final reduction
  eps_kernel<<<ew_blocks, blk, 0, stream>>>(E16);
  jvp_kernel<<<dim3(16, 32), blk, 0, stream>>>(Yt16, E16, Wt, U, jpart);
  finish_kernel<<<1, 256, 0, stream>>>(jpart, out_j);
}
